// Attention_18923625906513
// MI455X (gfx1250) — compile-verified, matrix operand order checked by reading, not run on hardware
//
#include <hip/hip_runtime.h>
#include <math.h>

// ---------------- WMMA types / helpers (gfx1250, wave32) ----------------
typedef __attribute__((ext_vector_type(16))) __bf16 v16bf;
typedef __attribute__((ext_vector_type(8)))  float  v8f;

union Frag {
  v16bf v;
  uint4 q[2];
  unsigned short u[16];
};

__device__ __forceinline__ unsigned short f2bf(float f) {
  unsigned u = __builtin_bit_cast(unsigned, f);
  unsigned r = (u + 0x7FFFu + ((u >> 16) & 1u)) >> 16;
  return (unsigned short)r;
}

__device__ __forceinline__ v16bf ld_frag(const unsigned short* p) {
  // p points at (row, k0 + (lane>=16 ? 8 : 0)); elems 0..7 = p[0..7], 8..15 = p[16..23]
  Frag f;
  f.q[0] = *(const uint4*)(p);
  f.q[1] = *(const uint4*)(p + 16);
  return f.v;
}

__device__ __forceinline__ v8f wmma_bf16(v16bf a, v16bf b, v8f c) {
  return __builtin_amdgcn_wmma_f32_16x16x32_bf16(false, a, false, b, (short)0, c, false, false);
}

// ---------------- small utility kernels ----------------
__global__ void cvt_kernel(const float* __restrict__ src, unsigned short* __restrict__ dst, long n) {
  for (long i = (long)blockIdx.x * blockDim.x + threadIdx.x; i < n; i += (long)gridDim.x * blockDim.x)
    dst[i] = f2bf(src[i]);
}

// src [K,N] fp32 (in,out) -> dst [N,K] bf16
__global__ void wt_kernel(const float* __restrict__ src, unsigned short* __restrict__ dst, int K, int N) {
  long tot = (long)K * N;
  for (long i = (long)blockIdx.x * blockDim.x + threadIdx.x; i < tot; i += (long)gridDim.x * blockDim.x) {
    int n = (int)(i / K), k = (int)(i % K);
    dst[i] = f2bf(src[(long)k * N + n]);
  }
}

// cb_conv_w [384,384,3,3] OIHW -> dst[co*3456 + (dy*3+dx)*384 + ci] bf16
__global__ void convw_kernel(const float* __restrict__ src, unsigned short* __restrict__ dst) {
  long tot = 384L * 3456L;
  for (long i = (long)blockIdx.x * blockDim.x + threadIdx.x; i < tot; i += (long)gridDim.x * blockDim.x) {
    int co = (int)(i / 3456); int rem = (int)(i % 3456);
    int blk = rem / 384, ci = rem % 384;
    int dy = blk / 3, dx = blk % 3;
    dst[i] = f2bf(src[((co * 384 + ci) * 3 + dy) * 3 + dx]);
  }
}

// 2x2 avgpool NHWC: in f32 [B, 2H, 2H, 384] -> out bf16 [B, H, H, 384]
__global__ __launch_bounds__(384) void pool_kernel(const float* __restrict__ in,
                                                   unsigned short* __restrict__ out, int Hout) {
  int pix = blockIdx.x;              // b*Hout*Hout + h*Hout + w
  int c = threadIdx.x;
  int HW2 = Hout * Hout;
  int bb = pix / HW2, pp = pix % HW2;
  int h = pp / Hout, w = pp % Hout;
  int Hin = Hout * 2;
  const float* p = in + ((long)(bb * Hin + 2 * h) * Hin + 2 * w) * 384 + c;
  float v = 0.25f * (p[0] + p[384] + p[(long)Hin * 384] + p[(long)Hin * 384 + 384]);
  out[(long)pix * 384 + c] = f2bf(v);
}

// ---------------- depthwise 3x3 conv + LayerNorm(C, eps 1e-6) -> bf16 ----------------
__global__ __launch_bounds__(384) void dwln_kernel(const float* __restrict__ x, const float* __restrict__ wdw,
                                                   const float* __restrict__ bdw, const float* __restrict__ g,
                                                   const float* __restrict__ b, unsigned short* __restrict__ out) {
  int pix = blockIdx.x;              // b*3136 + h*56 + w
  int c = threadIdx.x;
  int bb = pix / 3136, pp = pix % 3136;
  int h = pp / 56, w = pp % 56;
  float acc = bdw[c];
  for (int dy = 0; dy < 3; dy++) {
    int ih = h + dy - 1;
    if ((unsigned)ih >= 56u) continue;
    for (int dx = 0; dx < 3; dx++) {
      int iw = w + dx - 1;
      if ((unsigned)iw >= 56u) continue;
      acc += x[((long)(bb * 3136 + ih * 56 + iw)) * 384 + c] * wdw[c * 9 + dy * 3 + dx];
    }
  }
  __shared__ float sy[384];
  __shared__ float s1[128], s2[128];
  sy[c] = acc;
  __syncthreads();
  if (c < 128) {
    float a0 = sy[c], a1 = sy[c + 128], a2 = sy[c + 256];
    s1[c] = a0 + a1 + a2;
    s2[c] = a0 * a0 + a1 * a1 + a2 * a2;
  }
  __syncthreads();
  for (int s = 64; s > 0; s >>= 1) {
    if (c < s) { s1[c] += s1[c + s]; s2[c] += s2[c + s]; }
    __syncthreads();
  }
  float mean = s1[0] * (1.0f / 384.0f);
  float var  = s2[0] * (1.0f / 384.0f) - mean * mean;
  float rs = rsqrtf(var + 1e-6f);
  out[(long)pix * 384 + c] = f2bf((acc - mean) * rs * g[c] + b[c]);
}

// ---------------- fused MLP: h=GELU(ln@W1+b1); y=(h@W2+b2)*gamma + x ----------------
// 8 waves; hidden 16x1536 bf16 tile lives in LDS (48KB).
__global__ __launch_bounds__(256) void mlp_kernel(const unsigned short* __restrict__ LNin,
    const unsigned short* __restrict__ W1t, const float* __restrict__ b1,
    const unsigned short* __restrict__ W2t, const float* __restrict__ b2,
    const float* __restrict__ gamma, const float* __restrict__ xres, float* __restrict__ xlocal) {
  __shared__ __align__(16) unsigned short hid[16 * 1536];
  int lane = threadIdx.x & 31, wave = threadIdx.x >> 5;
  int m0 = blockIdx.x * 16;
  int l15 = lane & 15;
  int koff = (lane & 16) ? 8 : 0;      // K sub-offset for A/B fragments
  int rbase = koff;                    // D-row offset (+8 for upper half-wave)
  const unsigned short* pa = LNin + (long)(m0 + l15) * 384 + koff;
  // phase 1: 96 N-tiles of 16 over N=1536, 12 per wave
  for (int i = 0; i < 12; i++) {
    int n0 = (wave + 8 * i) * 16;
    const unsigned short* pb = W1t + (long)(n0 + l15) * 384 + koff;
    v8f acc = {};
    for (int k = 0; k < 384; k += 32)
      acc = wmma_bf16(ld_frag(pa + k), ld_frag(pb + k), acc);
    int col = n0 + l15;
    float bv = b1[col];
    for (int r = 0; r < 8; r++) {
      float v = acc[r] + bv;
      v = 0.5f * v * (1.0f + erff(v * 0.70710678118f));   // exact GELU
      hid[(rbase + r) * 1536 + col] = f2bf(v);
    }
  }
  __syncthreads();
  // phase 2: 24 N-tiles over N=384, 3 per wave, K=1536 from LDS
  const unsigned short* pa2 = &hid[l15 * 1536 + koff];
  for (int i = 0; i < 3; i++) {
    int n0 = (wave + 8 * i) * 16;
    const unsigned short* pb = W2t + (long)(n0 + l15) * 1536 + koff;
    v8f acc = {};
    for (int k = 0; k < 1536; k += 32)
      acc = wmma_bf16(ld_frag(pa2 + k), ld_frag(pb + k), acc);
    int col = n0 + l15;
    float bv = b2[col], gv = gamma[col];
    for (int r = 0; r < 8; r++) {
      long row = m0 + rbase + r;
      xlocal[row * 384 + col] = xres[row * 384 + col] + (acc[r] + bv) * gv;
    }
  }
}

// ---------------- 3x3 dense conv (implicit im2col, K=3456) + LN(1e-5) + ReLU ----------------
__global__ __launch_bounds__(256) void conv_kernel(const unsigned short* __restrict__ in,  // [16,28,28,384] bf16
    const unsigned short* __restrict__ Wt,  // [384,3456] bf16
    const float* __restrict__ g, const float* __restrict__ b, float* __restrict__ out) {
  __shared__ float cbuf[16][384];
  __shared__ float stats[16][2];
  int lane = threadIdx.x & 31, wave = threadIdx.x >> 5;
  int m0 = blockIdx.x * 16;
  int l15 = lane & 15;
  int p = m0 + l15;
  int bb = p / 784, pp = p % 784;
  int h = pp / 28, w = pp % 28;
  int koff = (lane & 16) ? 8 : 0;
  int rbase = koff;
  v8f acc0 = {}, acc1 = {}, acc2 = {};
  for (int k0 = 0; k0 < 3456; k0 += 32) {
    Frag fa;
    {
      int base = k0 + koff;
      int blk = base / 384, cin = base - blk * 384;
      int ih = h + blk / 3 - 1, iw = w + blk % 3 - 1;
      if ((unsigned)ih < 28u && (unsigned)iw < 28u)
        fa.q[0] = *(const uint4*)(in + ((long)(bb * 784 + ih * 28 + iw)) * 384 + cin);
      else
        fa.q[0] = make_uint4(0, 0, 0, 0);
    }
    {
      int base = k0 + koff + 16;
      int blk = base / 384, cin = base - blk * 384;
      int ih = h + blk / 3 - 1, iw = w + blk % 3 - 1;
      if ((unsigned)ih < 28u && (unsigned)iw < 28u)
        fa.q[1] = *(const uint4*)(in + ((long)(bb * 784 + ih * 28 + iw)) * 384 + cin);
      else
        fa.q[1] = make_uint4(0, 0, 0, 0);
    }
    int n0 = wave * 16;
    acc0 = wmma_bf16(fa.v, ld_frag(Wt + (long)(n0 + l15) * 3456 + koff + k0), acc0);
    acc1 = wmma_bf16(fa.v, ld_frag(Wt + (long)(n0 + 128 + l15) * 3456 + koff + k0), acc1);
    acc2 = wmma_bf16(fa.v, ld_frag(Wt + (long)(n0 + 256 + l15) * 3456 + koff + k0), acc2);
  }
  for (int r = 0; r < 8; r++) {
    cbuf[rbase + r][wave * 16 + l15]       = acc0[r];
    cbuf[rbase + r][wave * 16 + 128 + l15] = acc1[r];
    cbuf[rbase + r][wave * 16 + 256 + l15] = acc2[r];
  }
  __syncthreads();
  int t = threadIdx.x;
  if (t < 16) {
    float s = 0.f, ss = 0.f;
    for (int c = 0; c < 384; c++) { float v = cbuf[t][c]; s += v; ss += v * v; }
    float mean = s * (1.0f / 384.0f);
    float var = ss * (1.0f / 384.0f) - mean * mean;
    stats[t][0] = mean; stats[t][1] = rsqrtf(var + 1e-5f);
  }
  __syncthreads();
  for (int idx = t; idx < 16 * 384; idx += 256) {
    int row = idx / 384, c = idx % 384;
    float v = (cbuf[row][c] - stats[row][0]) * stats[row][1] * g[c] + b[c];
    out[(long)(m0 + row) * 384 + c] = fmaxf(v, 0.0f);
  }
}

// ---------------- generic WMMA GEMM: out[M,N] = A[M,K] @ Bt[N,K]^T (+bias) ----------------
// 4 waves, tile 16x64. M = gridDim.x*16, N = gridDim.y*64. K % 32 == 0.
__global__ __launch_bounds__(128) void gemm_kernel(const unsigned short* __restrict__ A,
    const unsigned short* __restrict__ Bt, const float* __restrict__ bias,
    void* __restrict__ out, int K, int N, int out_bf16) {
  int lane = threadIdx.x & 31, wave = threadIdx.x >> 5;
  int m0 = blockIdx.x * 16;
  int n0 = (blockIdx.y * 4 + wave) * 16;
  int l15 = lane & 15;
  int koff = (lane & 16) ? 8 : 0;
  int rbase = koff;
  const unsigned short* pa = A + (long)(m0 + l15) * K + koff;
  const unsigned short* pb = Bt + (long)(n0 + l15) * K + koff;
  v8f acc = {};
  for (int k = 0; k < K; k += 32)
    acc = wmma_bf16(ld_frag(pa + k), ld_frag(pb + k), acc);
  int col = n0 + l15;
  if (out_bf16) {
    unsigned short* o = (unsigned short*)out;
    for (int r = 0; r < 8; r++) o[(long)(m0 + rbase + r) * N + col] = f2bf(acc[r]);
  } else {
    float* o = (float*)out;
    float bv = bias ? bias[col] : 0.f;
    for (int r = 0; r < 8; r++) o[(long)(m0 + rbase + r) * N + col] = acc[r] + bv;
  }
}

// ---------------- single-wave MHA: softmax(Q K^T * 0.125) V, 6 heads x hd 64 ----------------
// mode: 0 = store f32, 1 = store bf16, 2 = accumulate f32 (+=)
#define NKP_MAX 224
__global__ __launch_bounds__(32) void mha_kernel(const unsigned short* __restrict__ qb,
    const unsigned short* __restrict__ kb, const unsigned short* __restrict__ vb,
    float* __restrict__ outf, unsigned short* __restrict__ outbf,
    int Nq, int Nk, int qStride, int qOff, int kvStride, int kOff, int vOff, int mode) {
  __shared__ __align__(16) unsigned short P[16 * NKP_MAX];
  int lane = threadIdx.x;
  int q0 = blockIdx.x * 16;
  int head = blockIdx.y;
  int bb = blockIdx.z;
  int l15 = lane & 15;
  int koff = (lane & 16) ? 8 : 0;
  int rbase = koff;
  int Nkp = (Nk + 31) & ~31;
  int nt = Nkp / 16;

  int qrow = q0 + l15; if (qrow >= Nq) qrow = Nq - 1;
  const unsigned short* qp = qb + (long)(bb * Nq + qrow) * qStride + qOff + head * 64 + koff;
  v16bf qa0 = ld_frag(qp);
  v16bf qa1 = ld_frag(qp + 32);

  v8f S[14];
  for (int j = 0; j < nt; j++) {
    int krow = j * 16 + l15; if (krow >= Nk) krow = Nk - 1;
    const unsigned short* kp = kb + (long)(bb * Nk + krow) * kvStride + kOff + head * 64 + koff;
    v8f a = {};
    a = wmma_bf16(qa0, ld_frag(kp), a);
    a = wmma_bf16(qa1, ld_frag(kp + 32), a);
    bool maskcol = (j * 16 + l15) >= Nk;
    for (int r = 0; r < 8; r++) S[j][r] = maskcol ? -1e30f : (a[r] * 0.125f);
  }
  // row-wise softmax (row's 16 cols live across one 16-lane half-wave)
  float inv[8];
  for (int r = 0; r < 8; r++) {
    float m = -1e30f;
    for (int j = 0; j < nt; j++) m = fmaxf(m, S[j][r]);
    for (int d = 1; d < 16; d <<= 1) m = fmaxf(m, __shfl_xor(m, d, 32));
    float s = 0.f;
    for (int j = 0; j < nt; j++) { float e = expf(S[j][r] - m); S[j][r] = e; s += e; }
    for (int d = 1; d < 16; d <<= 1) s += __shfl_xor(s, d, 32);
    inv[r] = 1.0f / s;
  }
  for (int j = 0; j < nt; j++) {
    int col = j * 16 + l15;
    for (int r = 0; r < 8; r++)
      P[(rbase + r) * NKP_MAX + col] = f2bf(S[j][r] * inv[r]);
  }
  __syncthreads();
  // O = P @ V : A from LDS (transposed layout), V fragments gathered
  const unsigned short* pp = &P[l15 * NKP_MAX + koff];
  for (int dt = 0; dt < 4; dt++) {
    int dd = dt * 16 + l15;
    v8f acc = {};
    for (int ks = 0; ks < Nkp; ks += 32) {
      v16bf pa = ld_frag(pp + ks);
      Frag fv;
      for (int i = 0; i < 8; i++) {
        int tok = ks + koff + i; if (tok >= Nk) tok = Nk - 1;
        fv.u[i] = vb[(long)(bb * Nk + tok) * kvStride + vOff + head * 64 + dd];
      }
      for (int i = 0; i < 8; i++) {
        int tok = ks + 16 + koff + i; if (tok >= Nk) tok = Nk - 1;
        fv.u[8 + i] = vb[(long)(bb * Nk + tok) * kvStride + vOff + head * 64 + dd];
      }
      acc = wmma_bf16(pa, fv.v, acc);
    }
    for (int r = 0; r < 8; r++) {
      int row = q0 + rbase + r;
      if (row < Nq) {
        long idx = (long)(bb * Nq + row) * 384 + head * 64 + dt * 16 + l15;
        if (mode == 0)      outf[idx] = acc[r];
        else if (mode == 1) outbf[idx] = f2bf(acc[r]);
        else                outf[idx] += acc[r];
      }
    }
  }
}

// ---------------- host ----------------
extern "C" void kernel_launch(void* const* d_in, const int* in_sizes, int n_in,
                              void* d_out, int out_size, void* d_ws, size_t ws_size,
                              hipStream_t stream) {
  const float* x        = (const float*)d_in[0];
  const float* gtok     = (const float*)d_in[1];
  const float* we_dw_w  = (const float*)d_in[2];
  const float* we_dw_b  = (const float*)d_in[3];
  const float* we_ln_g  = (const float*)d_in[4];
  const float* we_ln_b  = (const float*)d_in[5];
  const float* we_pw1_w = (const float*)d_in[6];
  const float* we_pw1_b = (const float*)d_in[7];
  const float* we_pw2_w = (const float*)d_in[8];
  const float* we_pw2_b = (const float*)d_in[9];
  const float* we_gamma = (const float*)d_in[10];
  const float* cb_conv_w= (const float*)d_in[11];
  const float* cb_ln_g  = (const float*)d_in[12];
  const float* cb_ln_b  = (const float*)d_in[13];
  const float* ga_qkv_w = (const float*)d_in[14];
  const float* ug_kv_w  = (const float*)d_in[15];
  const float* ug_q_w   = (const float*)d_in[16];
  const float* gb_kv_w  = (const float*)d_in[17];
  const float* gb_q_w   = (const float*)d_in[18];
  const float* proj_w   = (const float*)d_in[19];
  const float* proj_b   = (const float*)d_in[20];
  float* out = (float*)d_out;

  const long M1 = 16L * 3136;      // 50176 image tokens
  char* ws = (char*)d_ws;
  size_t cur = 0;
  auto alloc = [&](size_t bytes) -> void* {
    void* p = ws + cur;
    cur = (cur + bytes + 255) & ~(size_t)255;
    return p;
  };
  // weights (bf16, transposed [N,K])
  unsigned short* W1t    = (unsigned short*)alloc(1536L * 384 * 2);
  unsigned short* W2t    = (unsigned short*)alloc(384L * 1536 * 2);
  unsigned short* Wconvt = (unsigned short*)alloc(384L * 3456 * 2);
  unsigned short* Wqkvt  = (unsigned short*)alloc(1152L * 384 * 2);
  unsigned short* Wugkvt = (unsigned short*)alloc(768L * 384 * 2);
  unsigned short* Wugqt  = (unsigned short*)alloc(384L * 384 * 2);
  unsigned short* Wgbkvt = (unsigned short*)alloc(768L * 384 * 2);
  unsigned short* Wgbqt  = (unsigned short*)alloc(384L * 384 * 2);
  unsigned short* Wprojt = (unsigned short*)alloc(384L * 384 * 2);
  // activations
  unsigned short* xbf    = (unsigned short*)alloc(M1 * 384 * 2);
  unsigned short* lnbuf  = (unsigned short*)alloc(M1 * 384 * 2);  // reused as afinal
  float*          xlocal = (float*)alloc(M1 * 384 * 4);
  unsigned short* pooled1= (unsigned short*)alloc(16L * 784 * 384 * 2);
  float*          convout= (float*)alloc(16L * 784 * 384 * 4);
  unsigned short* xd     = (unsigned short*)alloc(16L * 196 * 384 * 2);
  unsigned short* qkv    = (unsigned short*)alloc(16L * 196 * 1152 * 2);
  unsigned short* xds    = (unsigned short*)alloc(16L * 196 * 384 * 2);
  unsigned short* gtokbf = (unsigned short*)alloc(784L * 384 * 2);
  unsigned short* qg     = (unsigned short*)alloc(784L * 384 * 2);
  unsigned short* kv2    = (unsigned short*)alloc(16L * 196 * 768 * 2);
  unsigned short* gtbf   = (unsigned short*)alloc(784L * 384 * 2);
  unsigned short* kv3    = (unsigned short*)alloc(784L * 768 * 2);
  unsigned short* qx     = (unsigned short*)alloc(M1 * 384 * 2);
  unsigned short* afinal = lnbuf;   // buf_ln is dead after mlp_kernel
  float* out_gt = out + M1 * 384;   // second tuple output

  // --- weight conversions ---
  wt_kernel<<<512, 256, 0, stream>>>(we_pw1_w, W1t, 384, 1536);
  wt_kernel<<<512, 256, 0, stream>>>(we_pw2_w, W2t, 1536, 384);
  convw_kernel<<<512, 256, 0, stream>>>(cb_conv_w, Wconvt);
  wt_kernel<<<512, 256, 0, stream>>>(ga_qkv_w, Wqkvt, 384, 1152);
  wt_kernel<<<512, 256, 0, stream>>>(ug_kv_w, Wugkvt, 384, 768);
  wt_kernel<<<256, 256, 0, stream>>>(ug_q_w, Wugqt, 384, 384);
  wt_kernel<<<512, 256, 0, stream>>>(gb_kv_w, Wgbkvt, 384, 768);
  wt_kernel<<<256, 256, 0, stream>>>(gb_q_w, Wgbqt, 384, 384);
  wt_kernel<<<256, 256, 0, stream>>>(proj_w, Wprojt, 384, 384);
  // --- activation conversions ---
  cvt_kernel<<<4096, 256, 0, stream>>>(x, xbf, M1 * 384);
  cvt_kernel<<<256, 256, 0, stream>>>(gtok, gtokbf, 784L * 384);

  // --- wsas[0]: dwconv + LN, then fused MLP with residual + layer scale ---
  dwln_kernel<<<50176, 384, 0, stream>>>(x, we_dw_w, we_dw_b, we_ln_g, we_ln_b, lnbuf);
  mlp_kernel<<<3136, 256, 0, stream>>>(lnbuf, W1t, we_pw1_b, W2t, we_pw2_b, we_gamma, x, xlocal);

  // --- down_1, wsas[1]: conv3x3 + LN + ReLU, down_2 ---
  pool_kernel<<<16 * 784, 384, 0, stream>>>(xlocal, pooled1, 28);
  conv_kernel<<<784, 256, 0, stream>>>(pooled1, Wconvt, cb_ln_g, cb_ln_b, convout);
  pool_kernel<<<16 * 196, 384, 0, stream>>>(convout, xd, 14);

  // --- global aggregation: qkv + MHSA over 196 tokens ---
  gemm_kernel<<<dim3(196, 18), 128, 0, stream>>>(xd, Wqkvt, nullptr, qkv, 384, 1152, 1);
  mha_kernel<<<dim3(13, 6, 16), 32, 0, stream>>>(qkv, qkv, qkv, nullptr, xds,
                                                 196, 196, 1152, 0, 1152, 384, 768, 1);
  // --- global token update: q = global tokens, kv from x_ds ---
  gemm_kernel<<<dim3(196, 12), 128, 0, stream>>>(xds, Wugkvt, nullptr, kv2, 384, 768, 1);
  gemm_kernel<<<dim3(49, 6), 128, 0, stream>>>(gtokbf, Wugqt, nullptr, qg, 384, 384, 1);
  mha_kernel<<<dim3(4, 6, 16), 32, 0, stream>>>(qg, kv2, kv2, out_gt, nullptr,
                                                49, 196, 384, 0, 768, 0, 384, 0);
  cvt_kernel<<<256, 256, 0, stream>>>(out_gt, gtbf, 784L * 384);
  // --- global broadcast: q = image tokens, kv from gt; accumulate into x_local ---
  gemm_kernel<<<dim3(49, 12), 128, 0, stream>>>(gtbf, Wgbkvt, nullptr, kv3, 384, 768, 1);
  gemm_kernel<<<dim3(3136, 6), 128, 0, stream>>>(xbf, Wgbqt, nullptr, qx, 384, 384, 1);
  mha_kernel<<<dim3(196, 6, 16), 32, 0, stream>>>(qx, kv3, kv3, xlocal, nullptr,
                                                  3136, 49, 384, 0, 768, 0, 384, 2);
  // --- final projection: (x_local + x_global) @ proj_w + proj_b ---
  cvt_kernel<<<4096, 256, 0, stream>>>(xlocal, afinal, M1 * 384);
  gemm_kernel<<<dim3(3136, 6), 128, 0, stream>>>(afinal, Wprojt, proj_b, out, 384, 384, 0);
}
